// StaticGraphGRUCell__36386962932008
// MI455X (gfx1250) — compile-verified, matrix operand order checked by reading, not run on hardware
//
#include <hip/hip_runtime.h>
#include <hip/hip_bf16.h>
#include <math.h>
#include <stdint.h>

// Problem constants (from reference): B=2048, N=48, D=H=128, 3H=384
#define B_DIM   2048
#define N_NODES 48
#define KDIM    128   // D == H == 128
#define TH      384   // 3*H
#define H_DIM   128

typedef _Float16 f16;
typedef __attribute__((ext_vector_type(16))) _Float16 v16h;
typedef __attribute__((ext_vector_type(8)))  float    v8f;

// --------------------------------------------------------------------------
// K index for element e of a 16-bit WMMA fragment (ISA 7.12.2, 16-bit A 16x32):
//   lane 0-15  (half=0): VGPR p<4 -> K=2p,2p+1 ; p>=4 -> K=16+2(p-4),+1
//   lane 16-31 (half=1): +8 on both K groups
// --------------------------------------------------------------------------
__device__ __forceinline__ int frag_k(int e, int half_) {
    int p  = e >> 1;
    int k0 = (p < 4 ? 2 * p : 16 + 2 * (p - 4)) + 8 * half_;
    return k0 + (e & 1);
}

// --------------------------------------------------------------------------
// CDNA5 async global->LDS copy (16B per lane), tracked by ASYNCcnt.
// LDS byte offset = low 32 bits of the flat pointer (ISA 10.2 aperture calc).
// --------------------------------------------------------------------------
__device__ __forceinline__ void async_copy_b128(void* ldsDst, const void* gsrc) {
    unsigned lds = (unsigned)(uintptr_t)ldsDst;
    unsigned long long ga = (unsigned long long)(uintptr_t)gsrc;
    asm volatile("global_load_async_to_lds_b128 %0, %1, off"
                 :: "v"(lds), "v"(ga) : "memory");
}
__device__ __forceinline__ void wait_asynccnt0() {
    asm volatile("s_wait_asynccnt 0x0" ::: "memory");
}

// ==========================================================================
// Kernel 1: per-node GEMM  res[z][b][n][o] = sum_k A[b][n][k]*W[n][o][k] + bias[n][o]
//   z=0: A=x,  W=weight_ih, bias=bias_ih   z=1: A=hx, W=weight_hh, bias=bias_hh
// Block: 256 threads (8 waves). Block tile: 32 b-rows x 64 outputs, full K=128 in LDS.
// Grid: x = (B/32)*(TH/64) = 64*6 = 384, y = N_NODES, z = 2
// ==========================================================================
__global__ __launch_bounds__(256)
void node_gemm_kernel(const float* __restrict__ x, const float* __restrict__ hx,
                      const float* __restrict__ w_ih, const float* __restrict__ w_hh,
                      const float* __restrict__ b_ih, const float* __restrict__ b_hh,
                      const int* __restrict__ node_types,
                      f16* __restrict__ resIH, f16* __restrict__ resHH) {
    // LDS staged directly in WMMA fragment order: [subtile][kstep][lane][elem]
    __shared__ __align__(32) f16 ldsA[2][4][32][16];   //  8 KB: 32 rows x 128 K
    __shared__ __align__(32) f16 ldsB[4][4][32][16];   // 16 KB: 128 K x 64 cols

    const int n     = blockIdx.y;
    const int z     = blockIdx.z;
    const int btile = blockIdx.x & 63;        // B/32 = 64 tiles
    const int otile = blockIdx.x >> 6;        // TH/64 = 6 tiles
    const int bbase = btile * 32;
    const int obase = otile * 64;
    const int wn    = node_types[n];

    const float* __restrict__ src  = (z == 0) ? x    : hx;
    const float* __restrict__ wsrc = (z == 0) ? w_ih : w_hh;
    const float* __restrict__ bsrc = (z == 0) ? b_ih : b_hh;
    f16* __restrict__ resout       = (z == 0) ? resIH : resHH;

    // ---- stage A (activations) : 256 threads cover 2*4*32 fragments ----
    {
        const int t     = threadIdx.x;
        const int m_sub = t >> 7;             // 0..1
        const int kstep = (t >> 5) & 3;       // 0..3
        const int lane  = t & 31;
        const int row   = bbase + m_sub * 16 + (lane & 15);
        const int half_ = lane >> 4;
        const float* ap = src + ((size_t)row * N_NODES + n) * KDIM;
        f16* dst = &ldsA[m_sub][kstep][lane][0];
#pragma unroll
        for (int e = 0; e < 16; ++e)
            dst[e] = (f16)ap[kstep * 32 + frag_k(e, half_)];
        // stream-ahead hint for the next b-tile of this node's activations
        if (row + 32 < B_DIM)
            __builtin_prefetch(src + ((size_t)(row + 32) * N_NODES + n) * KDIM, 0, 1);
    }

    // ---- stage B (weights, memory is [o][k] i.e. K-major per output) ----
    for (int q = threadIdx.x; q < 512; q += 256) {
        const int n_sub = q >> 7;             // 0..3
        const int kstep = (q >> 5) & 3;       // 0..3
        const int lane  = q & 31;
        const int o     = obase + n_sub * 16 + (lane & 15);
        const int half_ = lane >> 4;
        const float* wp = wsrc + ((size_t)wn * TH + o) * KDIM;
        f16* dst = &ldsB[n_sub][kstep][lane][0];
#pragma unroll
        for (int e = 0; e < 16; ++e)
            dst[e] = (f16)wp[kstep * 32 + frag_k(e, half_)];
    }

    __syncthreads();

    // ---- compute: wave w -> 16x16 C tile at (m_sub, n_sub) ----
    const int wave  = threadIdx.x >> 5;
    const int lane  = threadIdx.x & 31;
    const int m_sub = wave >> 2;              // 0..1
    const int n_sub = wave & 3;               // 0..3

    v8f acc = {};
#pragma unroll
    for (int kstep = 0; kstep < 4; ++kstep) {
        v16h a  = *(const v16h*)&ldsA[m_sub][kstep][lane][0];
        v16h bm = *(const v16h*)&ldsB[n_sub][kstep][lane][0];
        acc = __builtin_amdgcn_wmma_f32_16x16x32_f16(
            /*neg_a=*/false, a, /*neg_b=*/false, bm,
            /*c_mod=*/(short)0, acc, /*reuse_a=*/false, /*reuse_b=*/false);
    }

    // C/D layout: lanes 0-15 -> M=j, lanes 16-31 -> M=8+j ; N = lane&15
    const int o      = obase + n_sub * 16 + (lane & 15);
    const float bias = bsrc[(size_t)wn * TH + o];
    const int mhi    = (lane >> 4) * 8;
#pragma unroll
    for (int j = 0; j < 8; ++j) {
        const int brow = bbase + m_sub * 16 + mhi + j;
        resout[((size_t)brow * N_NODES + n) * TH + o] = (f16)(acc[j] + bias);
    }
}

// ==========================================================================
// Kernel 2: row L1-normalize G -> gx   (48x48, one small block)
// ==========================================================================
__global__ void norm_g_kernel(const float* __restrict__ G, float* __restrict__ gx) {
    const int r = threadIdx.x;
    if (r >= N_NODES) return;
    float s = 0.0f;
#pragma unroll
    for (int m = 0; m < N_NODES; ++m) s += fabsf(G[r * N_NODES + m]);
    const float inv = 1.0f / fmaxf(s, 1e-12f);
#pragma unroll
    for (int m = 0; m < N_NODES; ++m) gx[r * N_NODES + m] = G[r * N_NODES + m] * inv;
}

// ==========================================================================
// Kernel 3: graph mix + GRU gates + clockwork mask.
// Block handles one (b, half-of-H) pair. The 48-node slab of both f16
// intermediates for that h-range is pulled into LDS with CDNA5 async
// global->LDS b128 copies (ASYNCcnt), then the gx mix runs in fp32.
// Grid: 2*B blocks of 256 threads. LDS = 46 KB.
// ==========================================================================
__global__ __launch_bounds__(256)
void mix_gate_kernel(const f16* __restrict__ resIH, const f16* __restrict__ resHH,
                     const float* __restrict__ gx, const float* __restrict__ hx,
                     const int* __restrict__ tptr, float* __restrict__ out) {
    __shared__ __align__(16) float sGX[N_NODES][N_NODES];        // 9.2 KB
    __shared__ __align__(16) f16   sIH[N_NODES][192];            // 18.4 KB
    __shared__ __align__(16) f16   sHH[N_NODES][192];            // 18.4 KB

    const int b     = blockIdx.x >> 1;
    const int hbase = (blockIdx.x & 1) << 6;   // 0 or 64
    const int tid   = threadIdx.x;

    for (int i = tid; i < N_NODES * N_NODES; i += 256)
        ((float*)sGX)[i] = gx[i];

    // Async-copy 48 rows x 3 gate-segments x 128 bytes for each of the two
    // intermediate arrays: 2304 b128 chunks, 9 per thread.
    const f16* baseIH = resIH + (size_t)b * N_NODES * TH;
    const f16* baseHH = resHH + (size_t)b * N_NODES * TH;
    for (int c = tid; c < 2304; c += 256) {
        const int arr   = (c >= 1152);
        const int cc    = arr ? c - 1152 : c;
        const int mseg  = cc >> 3;             // 0..143
        const int chunk = cc & 7;              // 8 x 16B per 64-half segment
        const int m     = mseg / 3;
        const int seg   = mseg % 3;
        const f16* g    = (arr ? baseHH : baseIH)
                        + m * TH + seg * H_DIM + hbase + chunk * 8;
        f16* l          = (arr ? &sHH[0][0] : &sIH[0][0])
                        + m * 192 + seg * 64 + chunk * 8;
        async_copy_b128(l, g);
    }
    wait_asynccnt0();
    __syncthreads();

    const int tval = tptr[0];

    for (int idx = tid; idx < N_NODES * 64; idx += 256) {
        const int nn = idx >> 6;
        const int hl = idx & 63;
        const int h  = hbase + hl;

        float ir = 0.f, iz = 0.f, in_ = 0.f, hr = 0.f, hz = 0.f, hn = 0.f;
#pragma unroll 8
        for (int m = 0; m < N_NODES; ++m) {
            const float g = sGX[nn][m];
            ir  += g * (float)sIH[m][hl];
            iz  += g * (float)sIH[m][64 + hl];
            in_ += g * (float)sIH[m][128 + hl];
            hr  += g * (float)sHH[m][hl];
            hz  += g * (float)sHH[m][64 + hl];
            hn  += g * (float)sHH[m][128 + hl];
        }

        const float r  = 1.0f / (1.0f + __expf(-(ir + hr)));
        const float zg = 1.0f / (1.0f + __expf(-(iz + hz)));
        const float ng = tanhf(in_ + r * hn);
        const float hxv = hx[((size_t)b * N_NODES + nn) * H_DIM + h];
        float hy = ng - ng * zg + zg * hxv;

        const float phase = floorf((float)h * (8.0f / (float)(H_DIM - 1)) + 1.0f);
        const float cm = (fmodf((float)tval + 1.0f, phase) < 0.01f) ? 1.0f : 0.0f;
        out[((size_t)b * N_NODES + nn) * H_DIM + h] = cm * hy + (1.0f - cm) * hxv;
    }
}

// ==========================================================================
extern "C" void kernel_launch(void* const* d_in, const int* in_sizes, int n_in,
                              void* d_out, int out_size, void* d_ws, size_t ws_size,
                              hipStream_t stream) {
    const float* x     = (const float*)d_in[0];
    const float* hx    = (const float*)d_in[1];
    const float* G     = (const float*)d_in[2];
    // d_in[3] = G_add (unused by reference)
    const float* w_ih  = (const float*)d_in[4];
    const float* w_hh  = (const float*)d_in[5];
    const float* b_ih  = (const float*)d_in[6];
    const float* b_hh  = (const float*)d_in[7];
    const int* ntypes  = (const int*)d_in[8];
    const int* tptr    = (const int*)d_in[9];
    float* out         = (float*)d_out;

    const size_t resElems = (size_t)B_DIM * N_NODES * TH;   // per matrix
    f16*   resIH = (f16*)d_ws;
    f16*   resHH = resIH + resElems;
    float* gx    = (float*)((char*)d_ws + 2 * resElems * sizeof(f16));

    norm_g_kernel<<<1, 64, 0, stream>>>(G, gx);

    dim3 gridA((B_DIM / 32) * (TH / 64), N_NODES, 2);       // 384 x 48 x 2
    node_gemm_kernel<<<gridA, 256, 0, stream>>>(x, hx, w_ih, w_hh, b_ih, b_hh,
                                                ntypes, resIH, resHH);

    mix_gate_kernel<<<2 * B_DIM, 256, 0, stream>>>(resIH, resHH, gx, hx, tptr, out);
}
